// MultiHeadAttention_28621662061191
// MI455X (gfx1250) — compile-verified, matrix operand order checked by reading, not run on hardware
//
#include <hip/hip_runtime.h>
#include <hip/hip_bf16.h>

// Problem constants (match reference)
#define BB 2
#define SS 2048
#define EE 1024
#define HH 16
#define DH 64
#define HDH (HH * DH)   // 1024

typedef __bf16 bf16_t;
typedef bf16_t v16bf __attribute__((ext_vector_type(16)));
typedef bf16_t v8bf  __attribute__((ext_vector_type(8)));
typedef float  v8f   __attribute__((ext_vector_type(8)));
typedef uint32_t u32x4 __attribute__((ext_vector_type(4)));
typedef uint32_t u32x8 __attribute__((ext_vector_type(8)));

__device__ __forceinline__ v8f zero8() {
  v8f z = {0.f, 0.f, 0.f, 0.f, 0.f, 0.f, 0.f, 0.f};
  return z;
}

__device__ __forceinline__ v8f wmma_bf16(v16bf a, v16bf b, v8f c) {
  // D(f32 16x16) = A(16x32 bf16) * B(32x16 bf16) + C
  return __builtin_amdgcn_wmma_f32_16x16x32_bf16(
      false, a, false, b, (short)0, c, false, false);
}

// Load a 16x32 A-fragment (or a B-fragment supplied as B^T) from row-major
// bf16 memory (global or LDS). Per ISA 16-bit A layout:
//   lanes 0-15 : row = lane,    K = {k0+0..7,  k0+16..23}
//   lanes16-31 : row = lane-16, K = {k0+8..15, k0+24..31}
__device__ __forceinline__ v16bf load_frag(const bf16_t* base, int stride,
                                           int row0, int k0, int lane) {
  const bf16_t* p = base + (size_t)(row0 + (lane & 15)) * stride + k0 + ((lane >> 4) << 3);
  v8bf lo = *(const v8bf*)p;
  v8bf hi = *(const v8bf*)(p + 16);
  return __builtin_shufflevector(lo, hi, 0, 1, 2, 3, 4, 5, 6, 7,
                                 8, 9, 10, 11, 12, 13, 14, 15);
}

// ---- CDNA5 async data movers ------------------------------------------------
// Per-lane async copy global -> LDS (ASYNCcnt).
__device__ __forceinline__ void async_b128_to_lds(uint32_t lds_off, const void* gptr) {
  asm volatile("global_load_async_to_lds_b128 %0, %1, off"
               :: "v"(lds_off), "v"((unsigned long long)(uintptr_t)gptr)
               : "memory");
}
__device__ __forceinline__ void wait_async_le2() {
  asm volatile("s_wait_asynccnt 0x2" ::: "memory");
}
__device__ __forceinline__ void wait_async_0() {
  asm volatile("s_wait_asynccnt 0x0" ::: "memory");
}

// TDM: DMA one contiguous run of `nelem` bf16 from global into LDS (TENSORcnt).
// D# per CDNA5 ISA (1-line tile, data_size = 2 bytes).
__device__ __forceinline__ void tdm_load_row(uint32_t lds_off, const void* gptr,
                                             uint32_t nelem) {
  const unsigned long long ga = (unsigned long long)(uintptr_t)gptr;
  u32x4 g0;
  g0[0] = 1u;                                               // count=1 (valid user D#)
  g0[1] = lds_off;                                          // lds_addr
  g0[2] = (uint32_t)ga;                                     // global_addr lo
  g0[3] = (uint32_t)((ga >> 32) & 0x1FFFFFFu) | (2u << 30); // addr hi | type=2
  u32x8 g1;
  g1[0] = 1u << 16;                         // data_size=1 (2B), mask=0, no opts
  g1[1] = (nelem & 0xFFFFu) << 16;          // tensor_dim0[15:0]  @ bits[63:48]
  g1[2] = (nelem >> 16) | (1u << 16);       // tensor_dim0 hi, tensor_dim1=1
  g1[3] = (nelem & 0xFFFFu) << 16;          // tile_dim0 = nelem  @ bits[127:112]
  g1[4] = 1u;                               // tile_dim1=1, tile_dim2=0
  g1[5] = nelem;                            // tensor_dim0_stride lo
  g1[6] = 0u;
  g1[7] = 0u;
  asm volatile("tensor_load_to_lds %0, %1" :: "s"(g0), "s"(g1) : "memory");
}

// ---------------- conversion kernels ----------------
__global__ void k_f2bf(const float* __restrict__ src, bf16_t* __restrict__ dst, int n) {
  int i = blockIdx.x * blockDim.x + threadIdx.x;
  if (i < n) dst[i] = (bf16_t)src[i];
}

// W[H,E,DH] (fp32) -> W^T [H,DH,E] (bf16) for q,k,v weights
__global__ void k_wT(const float* __restrict__ Wq, const float* __restrict__ Wk,
                     const float* __restrict__ Wv, bf16_t* __restrict__ qT,
                     bf16_t* __restrict__ kT, bf16_t* __restrict__ vT) {
  int i = blockIdx.x * blockDim.x + threadIdx.x;
  if (i >= HH * EE * DH) return;
  int h = i / (EE * DH);
  int rem = i - h * (EE * DH);
  int e = rem / DH;
  int d = rem - e * DH;
  int o = (h * DH + d) * EE + e;
  qT[o] = (bf16_t)Wq[i];
  kT[o] = (bf16_t)Wk[i];
  vT[o] = (bf16_t)Wv[i];
}

// ---------------- QKV projection ----------------
// grid: (B*S/64, H), block 128 (4 waves). Wave w: rows m0..m0+15, head h.
// Outputs q,k as [B,H,S,DH] bf16, v transposed as [B,H,DH,S] bf16.
__global__ __launch_bounds__(128) void k_qkv(
    const bf16_t* __restrict__ xb, const bf16_t* __restrict__ wqT,
    const bf16_t* __restrict__ wkT, const bf16_t* __restrict__ wvT,
    const float* __restrict__ bq, const float* __restrict__ bk,
    const float* __restrict__ bv, bf16_t* __restrict__ qb,
    bf16_t* __restrict__ kb, bf16_t* __restrict__ vbT) {
  const int tid = threadIdx.x, w = tid >> 5, lane = tid & 31;
  const int r = lane & 15, sel = lane >> 4;
  const int h = blockIdx.y;
  const int m0 = blockIdx.x * 64 + w * 16;
  const bf16_t* wq = wqT + (size_t)h * DH * EE;
  const bf16_t* wk = wkT + (size_t)h * DH * EE;
  const bf16_t* wv = wvT + (size_t)h * DH * EE;

  v8f Aq[4] = {zero8(), zero8(), zero8(), zero8()};
  v8f Ak[4] = {zero8(), zero8(), zero8(), zero8()};
  v8f Av[4] = {zero8(), zero8(), zero8(), zero8()};

#pragma unroll 2
  for (int k = 0; k < EE; k += 32) {
    v16bf af = load_frag(xb, EE, m0, k, lane);
#pragma unroll
    for (int n = 0; n < 4; ++n) {
      Aq[n] = wmma_bf16(af, load_frag(wq, EE, n * 16, k, lane), Aq[n]);
      Ak[n] = wmma_bf16(af, load_frag(wk, EE, n * 16, k, lane), Ak[n]);
      Av[n] = wmma_bf16(af, load_frag(wv, EE, n * 16, k, lane), Av[n]);
    }
  }

#pragma unroll
  for (int n = 0; n < 4; ++n) {
    const int d = n * 16 + r;
    const float bqv = bq[h * DH + d];
    const float bkv = bk[h * DH + d];
    const float bvv = bv[h * DH + d];
#pragma unroll
    for (int g = 0; g < 8; ++g) {
      const int row = m0 + g + 8 * sel;      // in [0, B*S)
      const int b = row >> 11;               // / S
      const int s = row & (SS - 1);
      const size_t qi = (((size_t)(b * HH + h)) * SS + s) * DH + d;
      qb[qi] = (bf16_t)(Aq[n][g] + bqv);
      kb[qi] = (bf16_t)(Ak[n][g] + bkv);
      vbT[(((size_t)(b * HH + h)) * DH + d) * SS + s] = (bf16_t)(Av[n][g] + bvv);
    }
  }
}

// ---------------- flash attention ----------------
// grid: (S/64, B*H), block 128 (4 waves, 16 query rows each). K/V tiles are
// shared by all 4 waves and staged into LDS, double-buffered:
//   K tile [32 keys][64 d]  <- TDM  tensor_load_to_lds   (TENSORcnt)
//   V^T tile [64 d][32 keys]<- global_load_async_to_lds  (ASYNCcnt)
__global__ __launch_bounds__(128) void k_flash(
    const bf16_t* __restrict__ qb, const bf16_t* __restrict__ kb,
    const bf16_t* __restrict__ vbT, const int* __restrict__ mask,
    bf16_t* __restrict__ cb) {
  __shared__ bf16_t ktile[2][32][64];  // 2 x 4KB
  __shared__ bf16_t vtile[2][64][32];  // 2 x 4KB
  __shared__ bf16_t pbuf[4][16][40];   // per-wave P tile, row stride 80B

  const int tid = threadIdx.x, w = tid >> 5, lane = tid & 31;
  const int r = lane & 15, sel = lane >> 4;
  const int bh = blockIdx.y;
  const int b = bh >> 4, h = bh & 15;
  const int q0 = blockIdx.x * 64 + w * 16;

  const bf16_t* qh = qb + (size_t)bh * SS * DH;
  const bf16_t* kh = kb + (size_t)bh * SS * DH;
  const bf16_t* vh = vbT + (size_t)bh * DH * SS;
  const int* mrow = mask + (size_t)b * SS * SS;

  const v16bf qf0 = load_frag(qh, DH, q0, 0, lane);
  const v16bf qf1 = load_frag(qh, DH, q0, 32, lane);

  // stage tiles for key block t0 into buffer `bufi`
  auto issue_tiles = [&](int bufi, int t0) {
    if (w == 0)
      tdm_load_row((uint32_t)(size_t)&ktile[bufi][0][0],
                   kh + (size_t)t0 * DH, 32u * DH);
    const uint32_t vbase = (uint32_t)(size_t)&vtile[bufi][0][0];
#pragma unroll
    for (int cc = 0; cc < 2; ++cc) {
      const int c = tid + cc * 128;        // 256 chunks of 16B
      const int row = c >> 2, part = c & 3;
      async_b128_to_lds(vbase + (uint32_t)(row * 32 + part * 8) * 2,
                        vh + (size_t)row * SS + t0 + part * 8);
    }
  };

  float m[8], l[8];
#pragma unroll
  for (int g = 0; g < 8; ++g) { m[g] = -3.0e30f; l[g] = 0.f; }
  v8f O0 = zero8(), O1 = zero8(), O2 = zero8(), O3 = zero8();

  issue_tiles(0, 0);  // prologue

  const int nIt = SS / 32;
#pragma unroll 1
  for (int it = 0; it < nIt; ++it) {
    const int t0 = it * 32;
    const int buf = it & 1;
    if (it + 1 < nIt) {
      issue_tiles(buf ^ 1, t0 + 32);             // overlap DMA with compute
      if (w == 0) __builtin_amdgcn_s_wait_tensorcnt(1);
      wait_async_le2();                          // oldest batch (this buf) done
    } else {
      if (w == 0) __builtin_amdgcn_s_wait_tensorcnt(0);
      wait_async_0();
    }
    __syncthreads();  // tiles for `buf` visible to all waves

    const bf16_t* kt = &ktile[buf][0][0];
    const bf16_t* vt = &vtile[buf][0][0];

    // scores: two 16x16 tiles covering keys [t0, t0+32), K from LDS
    v8f s0 = zero8(), s1 = zero8();
    s0 = wmma_bf16(qf0, load_frag(kt, 64, 0, 0, lane), s0);
    s0 = wmma_bf16(qf1, load_frag(kt, 64, 0, 32, lane), s0);
    s1 = wmma_bf16(qf0, load_frag(kt, 64, 16, 0, lane), s1);
    s1 = wmma_bf16(qf1, load_frag(kt, 64, 16, 32, lane), s1);

    if (t0 + 32 < SS)
      __builtin_prefetch(mrow + (size_t)q0 * SS + t0 + 32, 0, 1);  // global_prefetch_b8

    // scale + mask (C layout: elem g -> row q0+g+8*sel, col = tile_col + r)
#pragma unroll
    for (int g = 0; g < 8; ++g) {
      const int qrow = q0 + g + 8 * sel;
      float v0 = s0[g] * 0.125f;  // 1/sqrt(64)
      float v1 = s1[g] * 0.125f;
      if (mrow[(size_t)qrow * SS + t0 + r] == 0) v0 = -1.0e9f;
      if (mrow[(size_t)qrow * SS + t0 + 16 + r] == 0) v1 = -1.0e9f;
      s0[g] = v0;
      s1[g] = v1;
    }

    // online softmax; each row lives in 16 adjacent lanes -> shfl_xor reductions
#pragma unroll
    for (int g = 0; g < 8; ++g) {
      float rm = fmaxf(s0[g], s1[g]);
      rm = fmaxf(rm, __shfl_xor(rm, 1, 32));
      rm = fmaxf(rm, __shfl_xor(rm, 2, 32));
      rm = fmaxf(rm, __shfl_xor(rm, 4, 32));
      rm = fmaxf(rm, __shfl_xor(rm, 8, 32));
      const float mn = fmaxf(m[g], rm);
      const float alpha = __expf(m[g] - mn);
      const float p0 = __expf(s0[g] - mn);
      const float p1 = __expf(s1[g] - mn);
      float ps = p0 + p1;
      ps += __shfl_xor(ps, 1, 32);
      ps += __shfl_xor(ps, 2, 32);
      ps += __shfl_xor(ps, 4, 32);
      ps += __shfl_xor(ps, 8, 32);
      l[g] = l[g] * alpha + ps;
      m[g] = mn;
      O0[g] *= alpha; O1[g] *= alpha; O2[g] *= alpha; O3[g] *= alpha;
      pbuf[w][g + 8 * sel][r] = (bf16_t)p0;
      pbuf[w][g + 8 * sel][16 + r] = (bf16_t)p1;
    }
    __syncthreads();  // C-layout -> A-fragment relayout through LDS

    v16bf pf;
    {
      const bf16_t* pr = &pbuf[w][r][0];
      v8bf lo = *(const v8bf*)(pr + sel * 8);
      v8bf hi = *(const v8bf*)(pr + 16 + sel * 8);
      pf = __builtin_shufflevector(lo, hi, 0, 1, 2, 3, 4, 5, 6, 7,
                                   8, 9, 10, 11, 12, 13, 14, 15);
    }
    // PV: B^T rows = V^T tile rows (d), K = 32 keys of this block, from LDS
    O0 = wmma_bf16(pf, load_frag(vt, 32, 0, 0, lane), O0);
    O1 = wmma_bf16(pf, load_frag(vt, 32, 16, 0, lane), O1);
    O2 = wmma_bf16(pf, load_frag(vt, 32, 32, 0, lane), O2);
    O3 = wmma_bf16(pf, load_frag(vt, 32, 48, 0, lane), O3);
    __syncthreads();  // all reads of `buf` done before its next DMA overwrite
  }

  // epilogue: normalize, write concat [B,S,H*DH] bf16 (head-major cols)
#pragma unroll
  for (int g = 0; g < 8; ++g) {
    const float inv = 1.0f / l[g];
    const int srow = q0 + g + 8 * sel;
    const size_t base = ((size_t)b * SS + srow) * HDH + h * DH + r;
    cb[base + 0]  = (bf16_t)(O0[g] * inv);
    cb[base + 16] = (bf16_t)(O1[g] * inv);
    cb[base + 32] = (bf16_t)(O2[g] * inv);
    cb[base + 48] = (bf16_t)(O3[g] * inv);
  }
}

// ---------------- output projection ----------------
// out[m, n] = sum_k cb[m, k] * Wo[n, k] + bo[n]   (Wo rows are already B^T rows)
// grid: (B*S/64, E/64), block 128; wave w: rows m0..m0+15, cols n0..n0+63.
__global__ __launch_bounds__(128) void k_oproj(
    const bf16_t* __restrict__ cb, const bf16_t* __restrict__ wob,
    const float* __restrict__ bo, float* __restrict__ out) {
  const int tid = threadIdx.x, w = tid >> 5, lane = tid & 31;
  const int r = lane & 15, sel = lane >> 4;
  const int m0 = blockIdx.x * 64 + w * 16;
  const int n0 = blockIdx.y * 64;

  v8f O[4] = {zero8(), zero8(), zero8(), zero8()};
#pragma unroll 2
  for (int k = 0; k < HDH; k += 32) {
    v16bf af = load_frag(cb, HDH, m0, k, lane);
#pragma unroll
    for (int j = 0; j < 4; ++j)
      O[j] = wmma_bf16(af, load_frag(wob, HDH, n0 + j * 16, k, lane), O[j]);
  }
#pragma unroll
  for (int j = 0; j < 4; ++j) {
    const float bv = bo[n0 + j * 16 + r];
#pragma unroll
    for (int g = 0; g < 8; ++g)
      out[(size_t)(m0 + g + 8 * sel) * EE + n0 + j * 16 + r] = O[j][g] + bv;
  }
}

// ---------------- launch ----------------
extern "C" void kernel_launch(void* const* d_in, const int* in_sizes, int n_in,
                              void* d_out, int out_size, void* d_ws, size_t ws_size,
                              hipStream_t stream) {
  const float* x  = (const float*)d_in[0];
  const int* mask = (const int*)d_in[1];
  const float* Wq = (const float*)d_in[2];
  const float* bq = (const float*)d_in[3];
  const float* Wk = (const float*)d_in[4];
  const float* bk = (const float*)d_in[5];
  const float* Wv = (const float*)d_in[6];
  const float* bv = (const float*)d_in[7];
  const float* Wo = (const float*)d_in[8];
  const float* bo = (const float*)d_in[9];
  float* out = (float*)d_out;

  // workspace carve (all offsets 256-B aligned)
  char* ws = (char*)d_ws;
  size_t off = 0;
  auto carve = [&](size_t bytes) {
    void* p = ws + off;
    off = (off + bytes + 255) & ~(size_t)255;
    return p;
  };
  bf16_t* xb  = (bf16_t*)carve((size_t)BB * SS * EE * 2);       // x, bf16
  bf16_t* wqT = (bf16_t*)carve((size_t)HH * DH * EE * 2);       // Wq^T [H,DH,E]
  bf16_t* wkT = (bf16_t*)carve((size_t)HH * DH * EE * 2);
  bf16_t* wvT = (bf16_t*)carve((size_t)HH * DH * EE * 2);
  bf16_t* wob = (bf16_t*)carve((size_t)EE * HDH * 2);           // Wo as-is (= B^T)
  bf16_t* qb  = (bf16_t*)carve((size_t)BB * HH * SS * DH * 2);  // [B,H,S,DH]
  bf16_t* kb  = (bf16_t*)carve((size_t)BB * HH * SS * DH * 2);  // [B,H,S,DH]
  bf16_t* vbT = (bf16_t*)carve((size_t)BB * HH * DH * SS * 2);  // [B,H,DH,S]
  bf16_t* cb  = (bf16_t*)carve((size_t)BB * SS * HDH * 2);      // concat [B,S,H*DH]
  (void)ws_size; (void)in_sizes; (void)n_in; (void)out_size;

  const int nX = BB * SS * EE;
  const int nWo = EE * HDH;
  const int nW = HH * EE * DH;

  k_f2bf<<<(nX + 255) / 256, 256, 0, stream>>>(x, xb, nX);
  k_f2bf<<<(nWo + 255) / 256, 256, 0, stream>>>(Wo, wob, nWo);
  k_wT<<<(nW + 255) / 256, 256, 0, stream>>>(Wq, Wk, Wv, wqT, wkT, wvT);

  k_qkv<<<dim3((BB * SS) / 64, HH), 128, 0, stream>>>(
      xb, wqT, wkT, wvT, bq, bk, bv, qb, kb, vbT);

  k_flash<<<dim3(SS / 64, BB * HH), 128, 0, stream>>>(qb, kb, vbT, mask, cb);

  k_oproj<<<dim3((BB * SS) / 64, EE / 64), 128, 0, stream>>>(cb, wob, bo, out);
}
